// AttentionHead_22162031247701
// MI455X (gfx1250) — matrix operand order repaired and checked by reading, not compiled, not run on hardware
//
#include <hip/hip_runtime.h>

typedef __attribute__((ext_vector_type(16))) __bf16 v16bf;
typedef __attribute__((ext_vector_type(8)))  float  v8f;

union FU { unsigned u[8]; v16bf v; };

__device__ __forceinline__ unsigned short f32_to_bf16(float f) {
    unsigned x = __float_as_uint(f);
    unsigned r = x + 0x7FFFu + ((x >> 16) & 1u);   // round-to-nearest-even
    return (unsigned short)(r >> 16);
}

// async copy global -> LDS (CDNA5), tracked by ASYNCcnt
__device__ __forceinline__ void async_g2l_b128(void* lds, const void* gaddr) {
    unsigned l = (unsigned)(size_t)lds;
    unsigned long long g = (unsigned long long)(size_t)gaddr;
    asm volatile("global_load_async_to_lds_b128 %0, %1, off" :: "v"(l), "v"(g) : "memory");
}
__device__ __forceinline__ void wait_async0() {
    asm volatile("s_wait_asynccnt 0" ::: "memory");
}

// ---------------------------------------------------------------------------
// Kernel 0: fp32 -> bf16 conversion (input and weight)
// ---------------------------------------------------------------------------
__global__ __launch_bounds__(256) void cvt_bf16(const float* __restrict__ src,
                                                unsigned short* __restrict__ dst,
                                                int n) {
    int i = (blockIdx.x * 256 + threadIdx.x) * 4;
    if (i + 3 < n) {
        float4 f = *(const float4*)(src + i);
        uint2 packed;
        packed.x = (unsigned)f32_to_bf16(f.x) | ((unsigned)f32_to_bf16(f.y) << 16);
        packed.y = (unsigned)f32_to_bf16(f.z) | ((unsigned)f32_to_bf16(f.w) << 16);
        *(uint2*)(dst + i) = packed;
    }
}

// ---------------------------------------------------------------------------
// Kernel 1: QKV GEMM  C[8192,3072] = A[8192,1024] * W[3072,1024]^T
// bf16 in/out, f32 WMMA acc. Block 256 thr (8 waves, 4m x 2n), tile 128x128,
// wave tile 32x64, K-step 32, double-buffered async global->LDS staging.
// ---------------------------------------------------------------------------
__global__ __launch_bounds__(256) void qkv_gemm(const unsigned short* __restrict__ A,
                                                const unsigned short* __restrict__ W,
                                                unsigned short* __restrict__ C) {
    __shared__ unsigned short As[2][128][40];   // 80B rows (16B aligned)
    __shared__ unsigned short Bs[2][128][40];

    const int tid  = threadIdx.x;
    const int lane = tid & 31, wid = tid >> 5;
    const int wm = wid & 3, wn = wid >> 2;      // 4 m-groups x 2 n-groups
    const int m0 = blockIdx.y * 128, n0 = blockIdx.x * 128;
    const int h = lane >> 4, l16 = lane & 15;

    v8f acc[2][4] = {};

    // stage one 128x32 K-slice of A and W into buffer `buf`
    auto load_tiles = [&](int kt, int buf) {
        #pragma unroll
        for (int it = 0; it < 2; ++it) {        // A: 128 rows x 4 b128
            int f = tid + 256 * it;
            int row = f >> 2, c = f & 3;
            async_g2l_b128(&As[buf][row][c * 8],
                           A + (size_t)(m0 + row) * 1024 + kt + c * 8);
        }
        #pragma unroll
        for (int it = 0; it < 2; ++it) {        // B: 128 rows x 4 b128
            int f = tid + 256 * it;
            int row = f >> 2, c = f & 3;
            async_g2l_b128(&Bs[buf][row][c * 8],
                           W + (size_t)(n0 + row) * 1024 + kt + c * 8);
        }
    };

    load_tiles(0, 0);
    wait_async0();
    __syncthreads();

    for (int kt = 0; kt < 1024; kt += 32) {
        const int cur = (kt >> 5) & 1;
        if (kt + 32 < 1024) load_tiles(kt + 32, cur ^ 1);   // overlap with WMMAs

        FU fa[2], fb[4];
        #pragma unroll
        for (int im = 0; im < 2; ++im) {
            int mr = wm * 32 + im * 16 + l16;
            #pragma unroll
            for (int vv = 0; vv < 4; ++vv) {
                fa[im].u[vv]     = *(const unsigned*)&As[cur][mr][h * 8 + 2 * vv];
                fa[im].u[vv + 4] = *(const unsigned*)&As[cur][mr][16 + h * 8 + 2 * vv];
            }
        }
        #pragma unroll
        for (int in = 0; in < 4; ++in) {
            int nr = wn * 64 + in * 16 + l16;
            #pragma unroll
            for (int vv = 0; vv < 4; ++vv) {
                fb[in].u[vv]     = *(const unsigned*)&Bs[cur][nr][h * 8 + 2 * vv];
                fb[in].u[vv + 4] = *(const unsigned*)&Bs[cur][nr][16 + h * 8 + 2 * vv];
            }
        }
        #pragma unroll
        for (int im = 0; im < 2; ++im)
            #pragma unroll
            for (int in = 0; in < 4; ++in)
                acc[im][in] = __builtin_amdgcn_wmma_f32_16x16x32_bf16(
                    false, fa[im].v, false, fb[in].v, (short)0, acc[im][in], false, false);

        wait_async0();      // own async writes to the alternate buffer done
        __syncthreads();    // everyone done reading `cur` / writing `cur^1`
    }

    #pragma unroll
    for (int im = 0; im < 2; ++im)
        #pragma unroll
        for (int in = 0; in < 4; ++in)
            #pragma unroll
            for (int vv = 0; vv < 8; ++vv) {
                int row = m0 + wm * 32 + im * 16 + vv + 8 * h;
                int col = n0 + wn * 64 + in * 16 + l16;
                C[(size_t)row * 3072 + col] = f32_to_bf16(acc[im][in][vv]);
            }
}

// ---------------------------------------------------------------------------
// Kernel 2: flash attention, reverse-causal mask (keep j >= i)
// out[b,i,:] = softmax_j( K_i . Q_j / 32 ) @ V
// block: 256 thr (8 waves), one 16-row i-tile; wave w owns e-slice [128w, 128w+128)
// ---------------------------------------------------------------------------
__global__ __launch_bounds__(256) void attn(const unsigned short* __restrict__ QKV,
                                            float* __restrict__ Out) {
    __shared__ unsigned short Ks[16][1032];      // K rows, 2064B stride (16B aligned)
    __shared__ unsigned short Vt[1024][40];      // V transposed: Vt[e][j], 80B stride
    __shared__ float Sred[8][16][32];            // per-wave partial scores
    __shared__ float Sfin[16][32];
    __shared__ unsigned short Pb[16][34];        // probs, bf16
    __shared__ float mrun[16], lrun[16], alph[16];
    __shared__ float pmax[2][16], psums[2][16];

    const int tid  = threadIdx.x;
    const int lane = tid & 31, wid = tid >> 5;
    const int h = lane >> 4, l16 = lane & 15;
    const int b  = blockIdx.x >> 7;
    const int i0 = (blockIdx.x & 127) * 16;
    const int base_b = b * 2048;
    const int base_i = base_b + i0;
    const int e0w = wid * 128;

    // ---- stage K rows: 16 x 1024 bf16 via async b128 copies ----
    #pragma unroll
    for (int it = 0; it < 8; ++it) {
        int f = tid + 256 * it;                  // 0..2047, 128 b128 per row
        int row = f >> 7, c = f & 127;
        async_g2l_b128(&Ks[row][c * 8],
                       QKV + (size_t)(base_i + row) * 3072 + 1024 + c * 8);
    }
    if (tid < 16) { mrun[tid] = -3.0e38f; lrun[tid] = 0.0f; }
    wait_async0();
    __syncthreads();

    v8f acc[8] = {};   // O accumulator: 16 rows x 128 cols

    const int jstart = i0 & ~31;
    for (int j0 = jstart; j0 < 2048; j0 += 32) {
        // ---- stage V tile transposed: micro-tile 8j x 4e per thread-iter ----
        #pragma unroll
        for (int it = 0; it < 4; ++it) {
            int tile = tid + 256 * it;           // 0..1023
            int jg8 = (tile & 3) * 8;
            int eg4 = (tile >> 2) * 4;
            uint2 rowv[8];
            #pragma unroll
            for (int jj = 0; jj < 8; ++jj)
                rowv[jj] = *(const uint2*)(QKV +
                    (size_t)(base_b + j0 + jg8 + jj) * 3072 + 2048 + eg4);
            #pragma unroll
            for (int q = 0; q < 4; ++q) {
                unsigned d[4];
                #pragma unroll
                for (int k = 0; k < 4; ++k) {
                    unsigned lo = (q < 2) ? (rowv[2 * k].x     >> (16 * q))       & 0xffffu
                                          : (rowv[2 * k].y     >> (16 * (q - 2))) & 0xffffu;
                    unsigned hi = (q < 2) ? (rowv[2 * k + 1].x >> (16 * q))       & 0xffffu
                                          : (rowv[2 * k + 1].y >> (16 * (q - 2))) & 0xffffu;
                    d[k] = lo | (hi << 16);
                }
                uint4 val; val.x = d[0]; val.y = d[1]; val.z = d[2]; val.w = d[3];
                *(uint4*)&Vt[eg4 + q][jg8] = val;
            }
        }
        // prefetch next V tile (speculative, L2)
        if (j0 + 32 < 2048)
            __builtin_prefetch(QKV + (size_t)(base_b + j0 + 32 + (tid & 31)) * 3072 + 2048, 0, 1);

        // ---- partial S (16 x 32) over this wave's 128-wide e-slice ----
        v8f sa[2] = {};
        #pragma unroll
        for (int ec = 0; ec < 4; ++ec) {
            int e0 = e0w + ec * 32;
            FU ka;
            #pragma unroll
            for (int vv = 0; vv < 4; ++vv) {
                ka.u[vv]     = *(const unsigned*)&Ks[l16][e0 + h * 8 + 2 * vv];
                ka.u[vv + 4] = *(const unsigned*)&Ks[l16][e0 + 16 + h * 8 + 2 * vv];
            }
            #pragma unroll
            for (int jh = 0; jh < 2; ++jh) {
                FU qb;
                int jr = j0 + jh * 16 + l16;
                const unsigned short* qp = QKV + (size_t)(base_b + jr) * 3072 + e0;
                #pragma unroll
                for (int vv = 0; vv < 4; ++vv) {
                    qb.u[vv]     = *(const unsigned*)(qp + h * 8 + 2 * vv);
                    qb.u[vv + 4] = *(const unsigned*)(qp + 16 + h * 8 + 2 * vv);
                }
                sa[jh] = __builtin_amdgcn_wmma_f32_16x16x32_bf16(
                    false, ka.v, false, qb.v, (short)0, sa[jh], false, false);
            }
        }
        #pragma unroll
        for (int jh = 0; jh < 2; ++jh)
            #pragma unroll
            for (int vv = 0; vv < 8; ++vv)
                Sred[wid][vv + 8 * h][jh * 16 + l16] = sa[jh][vv];
        __syncthreads();

        // ---- reduce partials across 8 waves, mask + scale ----
        #pragma unroll
        for (int r = 0; r < 2; ++r) {
            int ele = tid + 256 * r;
            int row = ele >> 5, col = ele & 31;
            float s = 0.0f;
            #pragma unroll
            for (int w = 0; w < 8; ++w) s += Sred[w][row][col];
            int jg = j0 + col, ig = i0 + row;
            Sfin[row][col] = (jg < ig) ? -1.0e30f : s * 0.03125f;
        }
        __syncthreads();

        // ---- online softmax (one wave: 2 lanes per row, wave-synchronous) ----
        if (tid < 32) {
            int row = tid & 15, hh = tid >> 4;
            int c0 = hh * 16;
            float lm = -3.0e38f;
            #pragma unroll
            for (int c = 0; c < 16; ++c) lm = fmaxf(lm, Sfin[row][c0 + c]);
            pmax[hh][row] = lm;
            float mold = mrun[row];
            float mx = fmaxf(mold, fmaxf(pmax[0][row], pmax[1][row]));
            float al = __expf(mold - mx);
            float ps = 0.0f;
            #pragma unroll
            for (int c = 0; c < 16; ++c) {
                float p = __expf(Sfin[row][c0 + c] - mx);
                ps += p;
                Pb[row][c0 + c] = f32_to_bf16(p);
            }
            psums[hh][row] = ps;
            if (hh == 0) {
                lrun[row] = lrun[row] * al + psums[0][row] + psums[1][row];
                mrun[row] = mx;
                alph[row] = al;
            }
        }
        __syncthreads();

        // ---- rescale O and accumulate P @ V on this wave's e-slice ----
        float al8[8];
        #pragma unroll
        for (int vv = 0; vv < 8; ++vv) al8[vv] = alph[vv + 8 * h];
        FU pf;
        #pragma unroll
        for (int vv = 0; vv < 4; ++vv) {
            pf.u[vv]     = *(const unsigned*)&Pb[l16][h * 8 + 2 * vv];
            pf.u[vv + 4] = *(const unsigned*)&Pb[l16][16 + h * 8 + 2 * vv];
        }
        #pragma unroll
        for (int t = 0; t < 8; ++t) {
            #pragma unroll
            for (int vv = 0; vv < 8; ++vv) acc[t][vv] *= al8[vv];
            FU vb;
            int ecol = e0w + t * 16 + l16;
            #pragma unroll
            for (int vv = 0; vv < 4; ++vv) {
                vb.u[vv]     = *(const unsigned*)&Vt[ecol][h * 8 + 2 * vv];
                vb.u[vv + 4] = *(const unsigned*)&Vt[ecol][16 + h * 8 + 2 * vv];
            }
            acc[t] = __builtin_amdgcn_wmma_f32_16x16x32_bf16(
                false, pf.v, false, vb.v, (short)0, acc[t], false, false);
        }
        __syncthreads();   // protect Vt / Sred / Pb before next iteration
    }

    // ---- normalize and store f32 output ----
    float li[8];
    #pragma unroll
    for (int vv = 0; vv < 8; ++vv) li[vv] = 1.0f / lrun[vv + 8 * h];
    #pragma unroll
    for (int t = 0; t < 8; ++t)
        #pragma unroll
        for (int vv = 0; vv < 8; ++vv) {
            int row = vv + 8 * h;
            int col = e0w + t * 16 + l16;
            Out[(size_t)(base_i + row) * 1024 + col] = acc[t][vv] * li[vv];
        }
}

// ---------------------------------------------------------------------------
extern "C" void kernel_launch(void* const* d_in, const int* in_sizes, int n_in,
                              void* d_out, int out_size, void* d_ws, size_t ws_size,
                              hipStream_t stream) {
    const float* inp = (const float*)d_in[0];   // [4,2048,1024]
    const float* w   = (const float*)d_in[1];   // [3072,1024]
    float* out       = (float*)d_out;           // [4,2048,1024]

    unsigned short* Abf = (unsigned short*)d_ws;            // 8192*1024 bf16
    unsigned short* Wbf = Abf + (size_t)8192 * 1024;        // 3072*1024 bf16
    unsigned short* QKV = Wbf + (size_t)3072 * 1024;        // 8192*3072 bf16

    const int nA = 8192 * 1024, nW = 3072 * 1024;
    cvt_bf16<<<nA / 4 / 256, 256, 0, stream>>>(inp, Abf, nA);
    cvt_bf16<<<nW / 4 / 256, 256, 0, stream>>>(w, Wbf, nW);

    dim3 g1(3072 / 128, 8192 / 128);
    qkv_gemm<<<g1, 256, 0, stream>>>(Abf, Wbf, QKV);

    attn<<<512, 256, 0, stream>>>(QKV, out);
}
